// HyperbolicGCNLayer_70360154243497
// MI455X (gfx1250) — matrix operand order repaired and checked by reading, not compiled, not run on hardware
//
#include <hip/hip_runtime.h>
#include <math.h>

// ---------------------------------------------------------------------------
// HyperbolicGCNLayer for MI455X (gfx1250, wave32).
//   h_raw = x @ W^T + b           -> WMMA f16 (f32 accum), 16x16x32 tiles,
//                                    W pre-converted to f16 once per launch,
//                                    x tile filled via async global->LDS,
//                                    32-node M blocks for 2x B-fragment reuse
//   h     = normalize(h_raw) * 1.5*sigmoid(x . mag_w + mag_b)
//   attention softmax segmented by tgt (atomic max / atomic add)
//   agg   = segment_sum(h_src * attn)
//   out   = clip(expmap0(h + agg), 0.95)
// ---------------------------------------------------------------------------

typedef __attribute__((ext_vector_type(16))) _Float16 v16h;
typedef __attribute__((ext_vector_type(8)))  float    v8f;
typedef __attribute__((ext_vector_type(4)))  float    v4f;
typedef __attribute__((ext_vector_type(4)))  int      v4i;

#define D_FEAT 256
#define MROWS  32                      // nodes per block (two 16-row M tiles)

#if __has_builtin(__builtin_amdgcn_global_load_async_to_lds_b128) && \
    __has_builtin(__builtin_amdgcn_s_wait_asynccnt)
#define HAVE_ASYNC_LDS 1
#else
#define HAVE_ASYNC_LDS 0
#endif

// monotone float<->int encoding so integer atomicMax == float max
__device__ __forceinline__ int f2ord(float f) {
  int i = __float_as_int(f);
  return i >= 0 ? i : (i ^ 0x7fffffff);
}
__device__ __forceinline__ float ord2f(int i) {
  return __int_as_float(i >= 0 ? i : (i ^ 0x7fffffff));
}

__device__ __forceinline__ float wave_sum(float v) {
#pragma unroll
  for (int off = 16; off; off >>= 1) v += __shfl_xor(v, off, 32);
  return v;
}

// --------------------------------------------------------------------------
// Kernel 0a: initialize workspace accumulators (harness does not re-zero ws)
// --------------------------------------------------------------------------
__global__ void k_init(float* __restrict__ agg, float* __restrict__ se,
                       int* __restrict__ mxenc, int n_nodes) {
  int stride = gridDim.x * blockDim.x;
  int i0 = blockIdx.x * blockDim.x + threadIdx.x;
  for (int i = i0; i < n_nodes * D_FEAT; i += stride) agg[i] = 0.0f;
  for (int i = i0; i < n_nodes; i += stride) {
    se[i] = 0.0f;
    mxenc[i] = f2ord(-1.0e9f);
  }
}

// --------------------------------------------------------------------------
// Kernel 0b: one-shot W f32 -> f16 conversion (shared by all GEMM blocks)
// --------------------------------------------------------------------------
__global__ void k_wconvert(const float* __restrict__ W, _Float16* __restrict__ Wh,
                           int n) {
  int i = blockIdx.x * blockDim.x + threadIdx.x;
  if (i * 4 < n) {
    v4f w = *(const v4f*)(W + i * 4);
    _Float16* o = Wh + i * 4;
    o[0] = (_Float16)w.x; o[1] = (_Float16)w.y;
    o[2] = (_Float16)w.z; o[3] = (_Float16)w.w;
  }
}

// --------------------------------------------------------------------------
// Kernel 1: node transform. One block = 32 nodes (2 M tiles). 256 thr = 8
// waves. Wave w owns output features [32w, 32w+31] (two 16-col N tiles); each
// B fragment is reused by both M tiles (4 WMMAs per wave per k-slab). Then
// the 8 waves split the 32 rows for norm / magnitude / attn projections.
// --------------------------------------------------------------------------
__global__ void __launch_bounds__(256)
k_node_transform(const float* __restrict__ x, const _Float16* __restrict__ Wh,
                 const float* __restrict__ b, const float* __restrict__ mag_w,
                 const float* __restrict__ mag_b, const float* __restrict__ attn_w,
                 float* __restrict__ h_out, float* __restrict__ alpha_s,
                 float* __restrict__ alpha_t, int n_nodes) {
  __shared__ float xsf[MROWS][D_FEAT + 8];   // x tile, f32 (8-bank skew)
  __shared__ float hs[MROWS][D_FEAT + 1];    // h_raw staging, f32

  const int tid = threadIdx.x;
  const int node0 = blockIdx.x * MROWS;

  // ---- fill x tile: async global->LDS (ASYNCcnt) or vector-load fallback
#if HAVE_ASYNC_LDS
#pragma unroll
  for (int it = 0; it < 8; ++it) {
    int i = it * 256 + tid;                  // float4 index, 2048 total
    int r = i >> 6, c = (i & 63) * 4;
    if (node0 + r < n_nodes) {
      __builtin_amdgcn_global_load_async_to_lds_b128(
          (__attribute__((address_space(1))) v4i*)
              (x + (size_t)(node0 + r) * D_FEAT + c),
          (__attribute__((address_space(3))) v4i*)&xsf[r][c],
          0, 0);
    }
  }
  __builtin_amdgcn_s_wait_asynccnt(0);
#else
  for (int i = tid; i < MROWS * 64; i += 256) {
    int r = i >> 6, c = (i & 63) * 4;
    if (node0 + r < n_nodes)
      *(v4f*)&xsf[r][c] = *(const v4f*)(x + (size_t)(node0 + r) * D_FEAT + c);
  }
#endif
  __syncthreads();

  const int wave = tid >> 5;
  const int lane = tid & 31;
  const int row  = lane & 15;   // A-matrix M row / B,C column-within-tile
  const int hi   = lane >> 4;   // lane half selector per ISA VGPR layouts

  const int f0 = wave * 32 + row;        // N-tile 0 feature column
  const int f1 = f0 + 16;                // N-tile 1 feature column
  v8f acc00 = {0.f, 0.f, 0.f, 0.f, 0.f, 0.f, 0.f, 0.f};  // Mtile0 x Ntile0
  v8f acc01 = acc00;                                       // Mtile0 x Ntile1
  v8f acc10 = acc00;                                       // Mtile1 x Ntile0
  v8f acc11 = acc00;                                       // Mtile1 x Ntile1

  for (int k0 = 0; k0 < D_FEAT; k0 += 32) {
    // A fragments (16x32 f16), one per M tile, each shared by both N tiles.
    // lane half hi -> K = k0 + hi*16 + j
    v16h a0, a1;
#pragma unroll
    for (int m = 0; m < 2; ++m) {
      const float* ab = &xsf[row + m * 16][k0 + hi * 16];
      v4f p0 = *(const v4f*)(ab);
      v4f p1 = *(const v4f*)(ab + 4);
      v4f p2 = *(const v4f*)(ab + 8);
      v4f p3 = *(const v4f*)(ab + 12);
      v16h a;
#pragma unroll
      for (int j = 0; j < 4; ++j) {
        a[j]      = (_Float16)p0[j];
        a[j + 4]  = (_Float16)p1[j];
        a[j + 8]  = (_Float16)p2[j];
        a[j + 12] = (_Float16)p3[j];
      }
      if (m == 0) a0 = a; else a1 = a;
    }
    // B (32x16 f16): B[k][n] = W[f][k]; direct 32B v16h loads from Wh,
    // each fragment reused by both M tiles.
    const _Float16* wb0 = Wh + (size_t)f0 * D_FEAT + k0 + hi * 16;
    const _Float16* wb1 = Wh + (size_t)f1 * D_FEAT + k0 + hi * 16;
    __builtin_prefetch(wb0 + 32, 0, 3);   // WGP-scope prefetch of next K slab
    v16h b0 = *(const v16h*)wb0;
    v16h b1 = *(const v16h*)wb1;

    acc00 = __builtin_amdgcn_wmma_f32_16x16x32_f16(false, a0, false, b0,
                                                   (short)0, acc00, false, false);
    acc01 = __builtin_amdgcn_wmma_f32_16x16x32_f16(false, a0, false, b1,
                                                   (short)0, acc01, false, false);
    acc10 = __builtin_amdgcn_wmma_f32_16x16x32_f16(false, a1, false, b0,
                                                   (short)0, acc10, false, false);
    acc11 = __builtin_amdgcn_wmma_f32_16x16x32_f16(false, a1, false, b1,
                                                   (short)0, acc11, false, false);
  }

  // C layout: lane<16 -> M = i, N = lane; lane>=16 -> M = 8+i, N = lane-16
  const float bias0 = b[f0];
  const float bias1 = b[f1];
#pragma unroll
  for (int i = 0; i < 8; ++i) {
    hs[i + hi * 8][f0]      = acc00[i] + bias0;
    hs[i + hi * 8][f1]      = acc01[i] + bias1;
    hs[16 + i + hi * 8][f0] = acc10[i] + bias0;
    hs[16 + i + hi * 8][f1] = acc11[i] + bias1;
  }
  __syncthreads();

  // row-wise epilogue: 8 waves x 4 rows; each lane covers 8 channels
  for (int r = wave; r < MROWS; r += 8) {
    if (node0 + r >= n_nodes) continue;
    float hv[8];
    float ss = 0.0f, xd = 0.0f;
#pragma unroll
    for (int j = 0; j < 8; ++j) {
      const int c = lane * 8 + j;
      float v = hs[r][c];
      hv[j] = v;
      ss += v * v;
      xd += xsf[r][c] * mag_w[c];
    }
    ss = wave_sum(ss);
    xd = wave_sum(xd);
    const float nrm = fmaxf(sqrtf(ss), 1e-12f);
    const float mag = 1.5f / (1.0f + __expf(-(xd + mag_b[0])));
    const float scl = mag / nrm;

    float as_ = 0.0f, at_ = 0.0f;
#pragma unroll
    for (int j = 0; j < 8; ++j) {
      const int c = lane * 8 + j;
      const float hvn = hv[j] * scl;
      h_out[(size_t)(node0 + r) * D_FEAT + c] = hvn;
      as_ += hvn * attn_w[c];
      at_ += hvn * attn_w[D_FEAT + c];
    }
    as_ = wave_sum(as_);
    at_ = wave_sum(at_);
    if (lane == 0) {
      alpha_s[node0 + r] = as_;
      alpha_t[node0 + r] = at_;
    }
  }
}

// --------------------------------------------------------------------------
// Kernel 2: per-edge score + segment max (ordered-int atomic max)
// --------------------------------------------------------------------------
__global__ void k_edge_max(const int* __restrict__ ei, const float* __restrict__ alpha_s,
                           const float* __restrict__ alpha_t, float* __restrict__ scores,
                           int* __restrict__ mxenc, int n_edges) {
  int e = blockIdx.x * blockDim.x + threadIdx.x;
  if (e >= n_edges) return;
  const int s = ei[e];
  const int t = ei[n_edges + e];
  const float sc = alpha_s[s] + alpha_t[t];
  scores[e] = sc;
  atomicMax(&mxenc[t], f2ord(sc));
}

// --------------------------------------------------------------------------
// Kernel 3: exp(score - mx[tgt]) + segment sum
// --------------------------------------------------------------------------
__global__ void k_edge_exp(const int* __restrict__ ei, float* __restrict__ scores,
                           const int* __restrict__ mxenc, float* __restrict__ se,
                           int n_edges) {
  int e = blockIdx.x * blockDim.x + threadIdx.x;
  if (e >= n_edges) return;
  const int t = ei[n_edges + e];
  const float ex = __expf(scores[e] - ord2f(mxenc[t]));
  scores[e] = ex;
  atomicAdd(&se[t], ex);
}

// --------------------------------------------------------------------------
// Kernel 4: weighted aggregation. One wave (32 lanes) per edge; lanes cover
// the 256 channels in 8 strided steps of global_atomic_add_f32 (L2-resident).
// --------------------------------------------------------------------------
__global__ void __launch_bounds__(256)
k_edge_agg(const int* __restrict__ ei, const float* __restrict__ h,
           const float* __restrict__ scores, const float* __restrict__ se,
           float* __restrict__ agg, int n_edges) {
  const int wv = (blockIdx.x * blockDim.x + threadIdx.x) >> 5;
  const int lane = threadIdx.x & 31;
  if (wv >= n_edges) return;
  const int s = ei[wv];
  const int t = ei[n_edges + wv];
  const float attn = scores[wv] / (se[t] + 1e-10f);
  const float* hrow = h + (size_t)s * D_FEAT;
  float* arow = agg + (size_t)t * D_FEAT;
#pragma unroll
  for (int j = 0; j < 8; ++j) {
    const int c = lane + j * 32;
    atomicAdd(&arow[c], hrow[c] * attn);
  }
}

// --------------------------------------------------------------------------
// Kernel 5: expmap0 + norm clip. One wave per node.
// --------------------------------------------------------------------------
__global__ void __launch_bounds__(256)
k_finalize(const float* __restrict__ h, const float* __restrict__ agg,
           float* __restrict__ out, int n_nodes) {
  const int wv = (blockIdx.x * blockDim.x + threadIdx.x) >> 5;
  const int lane = threadIdx.x & 31;
  if (wv >= n_nodes) return;
  float v[8];
  float ss = 0.0f;
#pragma unroll
  for (int j = 0; j < 8; ++j) {
    const int c = lane + j * 32;
    v[j] = h[(size_t)wv * D_FEAT + c] + agg[(size_t)wv * D_FEAT + c];
    ss += v[j] * v[j];
  }
  ss = wave_sum(ss);
  const float un  = sqrtf(ss);
  const float unc = fmaxf(un, 1e-15f);
  const float th  = tanhf(unc);
  float scale = th / unc;                // expmap0 scale
  const float pnorm = un * scale;        // ||h_poincare|| (== tanh(un) for un>0)
  if (pnorm > 0.95f) scale *= 0.95f / (pnorm + 1e-8f);
#pragma unroll
  for (int j = 0; j < 8; ++j) {
    const int c = lane + j * 32;
    out[(size_t)wv * D_FEAT + c] = v[j] * scale;
  }
}

// --------------------------------------------------------------------------
extern "C" void kernel_launch(void* const* d_in, const int* in_sizes, int n_in,
                              void* d_out, int out_size, void* d_ws, size_t ws_size,
                              hipStream_t stream) {
  const float* x      = (const float*)d_in[0];
  const int*   ei     = (const int*)d_in[1];   // (2, E): src row then tgt row
  const float* W      = (const float*)d_in[2];
  const float* b      = (const float*)d_in[3];
  const float* mag_w  = (const float*)d_in[4];
  const float* mag_b  = (const float*)d_in[5];
  const float* attn_w = (const float*)d_in[6];
  float* out = (float*)d_out;

  const int n_nodes = in_sizes[0] / D_FEAT;   // 10000
  const int n_edges = in_sizes[1] / 2;        // 320000
  const int n_w     = in_sizes[2];            // 65536

  // workspace carve-up (all section sizes are multiples of 32 bytes)
  char* ws = (char*)d_ws;
  float*    h       = (float*)ws;     ws += (size_t)n_nodes * D_FEAT * 4;
  float*    agg     = (float*)ws;     ws += (size_t)n_nodes * D_FEAT * 4;
  float*    alpha_s = (float*)ws;     ws += (size_t)n_nodes * 4;
  float*    alpha_t = (float*)ws;     ws += (size_t)n_nodes * 4;
  float*    se      = (float*)ws;     ws += (size_t)n_nodes * 4;
  int*      mxenc   = (int*)ws;       ws += (size_t)n_nodes * 4;
  float*    scores  = (float*)ws;     ws += (size_t)n_edges * 4;
  _Float16* Wh      = (_Float16*)ws;  ws += (size_t)n_w * 2;

  k_init<<<2048, 256, 0, stream>>>(agg, se, mxenc, n_nodes);
  k_wconvert<<<(n_w / 4 + 255) / 256, 256, 0, stream>>>(W, Wh, n_w);

  k_node_transform<<<(n_nodes + MROWS - 1) / MROWS, 256, 0, stream>>>(
      x, Wh, b, mag_w, mag_b, attn_w, h, alpha_s, alpha_t, n_nodes);

  const int eb = (n_edges + 255) / 256;
  k_edge_max<<<eb, 256, 0, stream>>>(ei, alpha_s, alpha_t, scores, mxenc, n_edges);
  k_edge_exp<<<eb, 256, 0, stream>>>(ei, scores, mxenc, se, n_edges);

  k_edge_agg<<<(n_edges * 32 + 255) / 256, 256, 0, stream>>>(ei, h, scores, se, agg, n_edges);

  k_finalize<<<(n_nodes * 32 + 255) / 256, 256, 0, stream>>>(h, agg, out, n_nodes);
}
